// Encoder_12257836662805
// MI455X (gfx1250) — compile-verified, matrix operand order checked by reading, not run on hardware
//
#include <hip/hip_runtime.h>
#include <hip/hip_bf16.h>

// ---------------------------------------------------------------------------
// Transformer encoder (B=4,S=1024,D=1024,H=16,DH=64,L=6,DFF=4096) for gfx1250.
// bf16 storage everywhere (fp32 accumulate in WMMA). Tensor Data Mover (TDM)
// stages all GEMM/attention tiles into LDS, double-buffered via TENSORcnt.
// Weights are stored transposed [N][K] so both A and B WMMA fragments are
// K-contiguous in LDS (coalesced ds_load_2addr_b32).
// ---------------------------------------------------------------------------

#define BB   4
#define SS   1024
#define DD   1024
#define HH   16
#define DHD  64
#define LL   6
#define DFFD 4096
#define MROWS (BB * SS)   // 4096

typedef __attribute__((ext_vector_type(8)))  float        v8f;
typedef __attribute__((ext_vector_type(16))) __bf16       v16bf;
typedef __attribute__((ext_vector_type(4)))  unsigned int tdm_v4u;
typedef __attribute__((ext_vector_type(8)))  int          tdm_v8i;
typedef __attribute__((ext_vector_type(4)))  int          tdm_v4i;

__device__ __forceinline__ unsigned short f2bf_bits(float f) {
    unsigned u = __builtin_bit_cast(unsigned, f);
    u += 0x7FFFu + ((u >> 16) & 1u);          // round-to-nearest-even
    return (unsigned short)(u >> 16);
}
__device__ __forceinline__ __bf16 bits2bf(unsigned short s) {
    return __builtin_bit_cast(__bf16, s);
}
__device__ __forceinline__ float bf2f(unsigned short s) {
    return __builtin_bit_cast(float, (unsigned)s << 16);
}

// The TDM writes LDS behind the compiler's back (descriptor carries a raw
// offset, not a pointer). Escape the LDS object and clobber memory so the
// compiler must treat its contents as modified and emit real ds_loads.
__device__ __forceinline__ void lds_publish(const void* p) {
    asm volatile("" : : "v"(p) : "memory");
}

// ---------------------------------------------------------------------------
// TDM: load a 2-D bf16 tile (tileD0 contiguous elems x tileD1 rows, row
// stride strideElems) from global into LDS at byte offset ldsOff.
// padInterval != 0 appends one dword of LDS padding every 2^(padInterval+3)
// bytes (set to exactly one row) -> LDS row pitch = tileD0 + 2 elements.
// D# layout per CDNA5 ISA 8.3/8.4. Wave-level op, tracked by TENSORcnt.
// ---------------------------------------------------------------------------
__device__ __forceinline__ void tdm_load_2d_bf16(
    unsigned ldsOff, const void* gp,
    unsigned tileD0, unsigned tileD1, unsigned strideElems,
    unsigned tensorD0, unsigned tensorD1, unsigned padInterval)
{
    unsigned long long ga = (unsigned long long)gp;
    tdm_v4u g0 = { 1u,                                   // count=1 (valid D#)
                   ldsOff,                               // lds_addr (bytes)
                   (unsigned)ga,                         // global_addr[31:0]
                   ((unsigned)(ga >> 32) & 0x01FFFFFFu)  // global_addr[56:32]
                     | 0x80000000u };                    // type=2 ("image")
    int w0 = (1 << 16);                                  // data_size=1 (2 B)
    if (padInterval)
        w0 |= (1 << 20) | ((int)padInterval << 22);      // pad_enable, amt=1 dw
    tdm_v8i g1 = {
        w0,
        (int)((tensorD0 & 0xFFFFu) << 16),               // tensor_dim0 lo16
        (int)((tensorD0 >> 16) | ((tensorD1 & 0xFFFFu) << 16)),
        (int)((tensorD1 >> 16) | (tileD0 << 16)),        // tile_dim0
        (int)(tileD1 & 0xFFFFu),                         // tile_dim1 (dim2=0)
        (int)strideElems,                                // tensor_dim0_stride
        0, 0 };
    tdm_v4i z4 = { 0, 0, 0, 0 };
    tdm_v8i z8 = { 0, 0, 0, 0, 0, 0, 0, 0 };
    __builtin_amdgcn_tensor_load_to_lds(g0, g1, z4, z4, z8, 0);
}

// ---------------------------------------------------------------------------
// Convert + transpose one layer-batched weight: src [L][R][C] fp32 ->
// dst [L][C][R] bf16.  32x32 LDS tile, coalesced loads and stores.
// ---------------------------------------------------------------------------
__global__ __launch_bounds__(256)
void cvt_tr_kernel(const float* __restrict__ src,
                   unsigned short* __restrict__ dst, int R, int C)
{
    __shared__ float tile[32][33];
    const size_t layerOff = (size_t)blockIdx.z * R * C;
    src += layerOff;
    dst += layerOff;
    const int c0 = blockIdx.x * 32;
    const int r0 = blockIdx.y * 32;
    const int tx = threadIdx.x & 31;
    const int ty = threadIdx.x >> 5;      // 0..7
#pragma unroll
    for (int i = 0; i < 4; ++i)
        tile[ty + i * 8][tx] = src[(size_t)(r0 + ty + i * 8) * C + c0 + tx];
    __syncthreads();
#pragma unroll
    for (int i = 0; i < 4; ++i)
        dst[(size_t)(c0 + ty + i * 8) * R + r0 + tx] =
            f2bf_bits(tile[tx][ty + i * 8]);
}

// ---------------------------------------------------------------------------
// Embedding + sinusoidal PE -> bf16 activations
// ---------------------------------------------------------------------------
__global__ __launch_bounds__(256)
void embed_kernel(const int* __restrict__ tok, const float* __restrict__ emb,
                  unsigned short* __restrict__ x)
{
    const int row  = blockIdx.x;              // b*S + s
    const int spos = row & (SS - 1);
    const int t    = tok[row];
    const int tid  = threadIdx.x;
    const float LN10000 = 9.210340371976184f;
#pragma unroll
    for (int i = 0; i < 4; ++i) {
        const int d = tid + i * 256;
        const float rate = __expf(-LN10000 * (float)(2 * (d >> 1)) * (1.0f / (float)DD));
        const float ang  = (float)spos * rate;
        const float pe   = (d & 1) ? __cosf(ang) : __sinf(ang);
        x[(size_t)row * DD + d] = f2bf_bits(emb[(size_t)t * DD + d] + pe);
    }
}

// ---------------------------------------------------------------------------
// bf16 GEMM: C[M,N] = A[M,K]*Bt[N,K]^T + bias (+res) (+relu), fp32 accum.
// A row-major [M][K]; B transposed [N][K] (weights). 128x128 block tile,
// K-tile 64, TDM double-buffered LDS (A and B tiles are the same shape).
// 8 waves, each owns 64x32 => 4x2 WMMA accumulators; 16 WMMA per K-tile.
// ---------------------------------------------------------------------------
#define TLD 66                          // LDS row pitch (64 + 2 pad)
#define TILE_E (128 * TLD)              // 8448 elems per tile

__global__ __launch_bounds__(256)
void gemm_bias_kernel(const unsigned short* __restrict__ A,
                      const unsigned short* __restrict__ Bt,
                      const float* __restrict__ bias,
                      const unsigned short* __restrict__ res,
                      unsigned short* __restrict__ C,
                      int M, int N, int K, int relu)
{
    // single shared object -> LDS offset 0; A0,A1,B0,B1
    __shared__ unsigned short smem[4 * TILE_E];

    const int tid    = threadIdx.x;
    const int lane   = tid & 31;
    const int wave   = tid >> 5;        // 0..7
    const int waveM  = wave >> 2;       // 0..1
    const int waveN  = wave & 3;        // 0..3
    const int blockM = blockIdx.y * 128;
    const int blockN = blockIdx.x * 128;
    const int grp    = lane >> 4;
    const int lan    = lane & 15;
    const int kOff   = grp * 8;

    const unsigned aOffB[2] = { 0u, (unsigned)(TILE_E * 2) };
    const unsigned bOffB[2] = { (unsigned)(2 * TILE_E * 2),
                                (unsigned)(3 * TILE_E * 2) };

    const v8f vzero = {0.f,0.f,0.f,0.f,0.f,0.f,0.f,0.f};
    v8f acc[4][2];
#pragma unroll
    for (int ms = 0; ms < 4; ++ms)
#pragma unroll
        for (int ns = 0; ns < 2; ++ns) acc[ms][ns] = vzero;

    const int nT = K >> 6;              // K-tiles of 64
    if (wave == 0) {                    // prefetch tile 0
        tdm_load_2d_bf16(aOffB[0], A  + (size_t)blockM * K, 64, 128, K,
                         K, M - blockM, 4);
        tdm_load_2d_bf16(bOffB[0], Bt + (size_t)blockN * K, 64, 128, K,
                         K, N - blockN, 4);
    }

    for (int t = 0; t < nT; ++t) {
        const int cur = t & 1;
        if (wave == 0) {
            if (t + 1 < nT) {           // issue next tile, then retire current
                const int k0 = (t + 1) << 6;
                tdm_load_2d_bf16(aOffB[cur ^ 1], A  + (size_t)blockM * K + k0,
                                 64, 128, K, K - k0, M - blockM, 4);
                tdm_load_2d_bf16(bOffB[cur ^ 1], Bt + (size_t)blockN * K + k0,
                                 64, 128, K, K - k0, N - blockN, 4);
                __builtin_amdgcn_s_wait_tensorcnt(2);
            } else {
                __builtin_amdgcn_s_wait_tensorcnt(0);
            }
        }
        __syncthreads();
        lds_publish(smem);             // TDM modified LDS: force fresh ds_loads

        const unsigned short* As = &smem[cur * TILE_E];
        const unsigned short* Bs = &smem[(2 + cur) * TILE_E];

#pragma unroll
        for (int ks = 0; ks < 2; ++ks) {            // two 32-deep substeps
            const int kb = ks * 32;
            v16bf af[4];
#pragma unroll
            for (int ms = 0; ms < 4; ++ms) {
                const int row = waveM * 64 + ms * 16 + lan;
#pragma unroll
                for (int hh = 0; hh < 16; ++hh) {
                    const int kk = (hh & 7) + ((hh >> 3) << 4) + kOff;
                    af[ms][hh] = bits2bf(As[row * TLD + kb + kk]);
                }
            }
            v16bf bfr[2];
#pragma unroll
            for (int ns = 0; ns < 2; ++ns) {
                const int col = waveN * 32 + ns * 16 + lan;
#pragma unroll
                for (int hh = 0; hh < 16; ++hh) {
                    const int kk = (hh & 7) + ((hh >> 3) << 4) + kOff;
                    bfr[ns][hh] = bits2bf(Bs[col * TLD + kb + kk]);
                }
            }
#pragma unroll
            for (int ms = 0; ms < 4; ++ms)
#pragma unroll
                for (int ns = 0; ns < 2; ++ns)
                    acc[ms][ns] = __builtin_amdgcn_wmma_f32_16x16x32_bf16(
                        false, af[ms], false, bfr[ns], (short)0, acc[ms][ns],
                        false, false);
        }
        __syncthreads();
    }

#pragma unroll
    for (int ms = 0; ms < 4; ++ms) {
#pragma unroll
        for (int ns = 0; ns < 2; ++ns) {
            const int nG = blockN + waveN * 32 + ns * 16 + lan;
            const float bv = bias[nG];
#pragma unroll
            for (int r = 0; r < 8; ++r) {
                const int mG = blockM + waveM * 64 + ms * 16 + r + 8 * grp;
                float v = acc[ms][ns][r] + bv;
                if (res)  v += bf2f(res[(size_t)mG * N + nG]);
                if (relu) v = fmaxf(v, 0.0f);
                C[(size_t)mG * N + nG] = f2bf_bits(v);
            }
        }
    }
}

// ---------------------------------------------------------------------------
// Flash attention: workgroup = (128 queries, one (b,h)); 8 waves x 16 rows.
// K/V 32x64 blocks staged once per WG via TDM; scores + P*V via WMMA;
// online softmax with 16-lane shfl reductions; P re-laid out through LDS.
// ---------------------------------------------------------------------------
#define KV_LD 66                       // 64 + 2 pad
#define KT_E  (32 * KV_LD)             // 2112 elems
#define P_E   (16 * 34)                // per-wave P tile elems

__global__ __launch_bounds__(256)
void attention_kernel(const unsigned short* __restrict__ qb,
                      const unsigned short* __restrict__ kb,
                      const unsigned short* __restrict__ vb,
                      const float* __restrict__ mask,
                      unsigned short* __restrict__ ob)
{
    // single shared object -> LDS offset 0; [Ktile][Vtile][P x 8 waves]
    __shared__ unsigned short att_lds[2 * KT_E + 8 * P_E];

    const int tid  = threadIdx.x;
    const int lane = tid & 31;
    const int wave = tid >> 5;
    const int b    = blockIdx.z;
    const int h    = blockIdx.y;
    const int qBase= blockIdx.x * 128 + wave * 16;
    const int grp  = lane >> 4;
    const int lan  = lane & 15;
    const int kOff = grp * 8;

    const unsigned short* Kt = &att_lds[0];
    const unsigned short* Vt = &att_lds[KT_E];
    unsigned short*       Pw = &att_lds[2 * KT_E + wave * P_E];

    // Q fragments (pre-scaled by 1/sqrt(64)), reused across all key blocks
    const unsigned short* qp = qb + ((size_t)b * SS + qBase) * DD + h * DHD;
    v16bf qa[2];
#pragma unroll
    for (int s = 0; s < 2; ++s)
#pragma unroll
        for (int hh = 0; hh < 16; ++hh) {
            const int dd = (hh & 7) + ((hh >> 3) << 4) + kOff + s * 32;
            qa[s][hh] = bits2bf(f2bf_bits(bf2f(qp[(size_t)lan * DD + dd]) * 0.125f));
        }

    const v8f vzero = {0.f,0.f,0.f,0.f,0.f,0.f,0.f,0.f};
    float mI[8], lI[8];
    v8f ctx[4];
#pragma unroll
    for (int r = 0; r < 8; ++r) { mI[r] = -1e30f; lI[r] = 0.0f; }
#pragma unroll
    for (int ns = 0; ns < 4; ++ns) ctx[ns] = vzero;

    for (int kb0 = 0; kb0 < SS; kb0 += 32) {
        if (wave == 0) {               // stage K,V blocks for the whole WG
            tdm_load_2d_bf16(0u, kb + ((size_t)b * SS + kb0) * DD + h * DHD,
                             64, 32, DD, DD, 32, 4);
            tdm_load_2d_bf16((unsigned)(KT_E * 2),
                             vb + ((size_t)b * SS + kb0) * DD + h * DHD,
                             64, 32, DD, DD, 32, 4);
            __builtin_amdgcn_s_wait_tensorcnt(0);
        }
        __syncthreads();
        lds_publish(att_lds);          // TDM modified LDS: force fresh ds_loads

        // ---- scores S(16x32) as two 16x16 C tiles ----
        v8f sAcc[2];
#pragma unroll
        for (int t = 0; t < 2; ++t) {
            const int key = t * 16 + lan;          // key row within block
            v16bf bk0, bk1;
#pragma unroll
            for (int hh = 0; hh < 16; ++hh) {
                const int dd = (hh & 7) + ((hh >> 3) << 4) + kOff;
                bk0[hh] = bits2bf(Kt[key * KV_LD + dd]);
                bk1[hh] = bits2bf(Kt[key * KV_LD + dd + 32]);
            }
            v8f s = __builtin_amdgcn_wmma_f32_16x16x32_bf16(
                false, qa[0], false, bk0, (short)0, vzero, false, false);
            sAcc[t] = __builtin_amdgcn_wmma_f32_16x16x32_bf16(
                false, qa[1], false, bk1, (short)0, s, false, false);
        }

        const float mk0 = mask[(size_t)b * SS + kb0 + lan]      * -1e9f;
        const float mk1 = mask[(size_t)b * SS + kb0 + 16 + lan] * -1e9f;

        // ---- online softmax (rows r + 8*grp, replicated across 16 lanes) ----
        float p0[8], p1[8];
#pragma unroll
        for (int r = 0; r < 8; ++r) {
            const float s0 = sAcc[0][r] + mk0;
            const float s1 = sAcc[1][r] + mk1;
            float mx = fmaxf(s0, s1);
#pragma unroll
            for (int off = 1; off < 16; off <<= 1)
                mx = fmaxf(mx, __shfl_xor(mx, off, 32));
            const float mnew  = fmaxf(mI[r], mx);
            const float alpha = __expf(mI[r] - mnew);
            const float e0 = __expf(s0 - mnew);
            const float e1 = __expf(s1 - mnew);
            float sum = e0 + e1;
#pragma unroll
            for (int off = 1; off < 16; off <<= 1)
                sum += __shfl_xor(sum, off, 32);
            lI[r] = lI[r] * alpha + sum;
            mI[r] = mnew;
            p0[r] = e0; p1[r] = e1;
#pragma unroll
            for (int ns = 0; ns < 4; ++ns) ctx[ns][r] *= alpha;
        }

        // ---- P: C-layout -> per-wave LDS -> A-fragment layout ----
#pragma unroll
        for (int r = 0; r < 8; ++r) {
            const int row = r + 8 * grp;
            Pw[row * 34 + lan]      = f2bf_bits(p0[r]);
            Pw[row * 34 + 16 + lan] = f2bf_bits(p1[r]);
        }
        __asm__ volatile("s_wait_dscnt 0" ::: "memory");  // per-wave LDS RAW

        v16bf pa;
#pragma unroll
        for (int hh = 0; hh < 16; ++hh) {
            const int kk = (hh & 7) + ((hh >> 3) << 4) + kOff;
            pa[hh] = bits2bf(Pw[lan * 34 + kk]);
        }

        // ---- ctx += P(16x32) * V(32x64) ----
#pragma unroll
        for (int ns = 0; ns < 4; ++ns) {
            v16bf vf;
#pragma unroll
            for (int hh = 0; hh < 16; ++hh) {
                const int kk = (hh & 7) + ((hh >> 3) << 4) + kOff;
                vf[hh] = bits2bf(Vt[kk * KV_LD + ns * 16 + lan]);
            }
            ctx[ns] = __builtin_amdgcn_wmma_f32_16x16x32_bf16(
                false, pa, false, vf, (short)0, ctx[ns], false, false);
        }
        __syncthreads();               // K/V tiles reused next iteration
    }

#pragma unroll
    for (int ns = 0; ns < 4; ++ns)
#pragma unroll
        for (int r = 0; r < 8; ++r) {
            const int m = r + 8 * grp;
            ob[((size_t)b * SS + qBase + m) * DD + h * DHD + ns * 16 + lan] =
                f2bf_bits(ctx[ns][r] * (1.0f / lI[r]));
        }
}

// ---------------------------------------------------------------------------
// Row LayerNorm over D=1024 (bf16 in; bf16 out, or fp32 out if yf != null)
// ---------------------------------------------------------------------------
__global__ __launch_bounds__(256)
void layernorm_kernel(const unsigned short* __restrict__ xin,
                      const float* __restrict__ g, const float* __restrict__ bt,
                      unsigned short* __restrict__ yb, float* __restrict__ yf)
{
    __shared__ float redA[8], redB[8];
    const int row  = blockIdx.x;
    const int tid  = threadIdx.x;
    const int lane = tid & 31, wave = tid >> 5;
    const unsigned short* xr = xin + (size_t)row * DD;

    float v[4]; float s = 0.0f;
#pragma unroll
    for (int i = 0; i < 4; ++i) { v[i] = bf2f(xr[tid + i * 256]); s += v[i]; }
#pragma unroll
    for (int off = 16; off; off >>= 1) s += __shfl_xor(s, off, 32);
    if (lane == 0) redA[wave] = s;
    __syncthreads();
    float mean = 0.0f;
#pragma unroll
    for (int w = 0; w < 8; ++w) mean += redA[w];
    mean *= (1.0f / (float)DD);

    float vs = 0.0f;
#pragma unroll
    for (int i = 0; i < 4; ++i) { const float d = v[i] - mean; vs += d * d; }
#pragma unroll
    for (int off = 16; off; off >>= 1) vs += __shfl_xor(vs, off, 32);
    if (lane == 0) redB[wave] = vs;
    __syncthreads();
    float var = 0.0f;
#pragma unroll
    for (int w = 0; w < 8; ++w) var += redB[w];
    var *= (1.0f / (float)DD);
    const float inv = rsqrtf(var + 1e-6f);

#pragma unroll
    for (int i = 0; i < 4; ++i) {
        const int d = tid + i * 256;
        const float o = (v[i] - mean) * inv * g[d] + bt[d];
        if (yf) yf[(size_t)row * DD + d] = o;
        else    yb[(size_t)row * DD + d] = f2bf_bits(o);
    }
}

// ---------------------------------------------------------------------------
// Host orchestration
// ---------------------------------------------------------------------------
static void launch_gemm(const unsigned short* A, const unsigned short* Bt,
                        const float* bias, const unsigned short* res,
                        unsigned short* C, int M, int N, int K, int relu,
                        hipStream_t stream)
{
    dim3 grid(N / 128, M / 128);
    gemm_bias_kernel<<<grid, 256, 0, stream>>>(A, Bt, bias, res, C, M, N, K, relu);
}

extern "C" void kernel_launch(void* const* d_in, const int* in_sizes, int n_in,
                              void* d_out, int out_size, void* d_ws, size_t ws_size,
                              hipStream_t stream)
{
    (void)in_sizes; (void)n_in; (void)out_size; (void)ws_size;

    const int*   tokens = (const int*)  d_in[0];
    const float* mask   = (const float*)d_in[1];
    const float* emb    = (const float*)d_in[2];
    const float* Wq     = (const float*)d_in[3];
    const float* bq     = (const float*)d_in[4];
    const float* Wk     = (const float*)d_in[5];
    const float* bk     = (const float*)d_in[6];
    const float* Wv     = (const float*)d_in[7];
    const float* bv     = (const float*)d_in[8];
    const float* Wo     = (const float*)d_in[9];
    const float* bo     = (const float*)d_in[10];
    const float* W1     = (const float*)d_in[11];
    const float* b1     = (const float*)d_in[12];
    const float* W2     = (const float*)d_in[13];
    const float* b2     = (const float*)d_in[14];
    const float* g1     = (const float*)d_in[15];
    const float* beta1  = (const float*)d_in[16];
    const float* g2     = (const float*)d_in[17];
    const float* beta2  = (const float*)d_in[18];

    typedef unsigned short us;
    char* ws = (char*)d_ws;
    size_t off = 0;
    auto take = [&](size_t elems) { us* p = (us*)(ws + off);
                                    off += ((elems * 2) + 255) & ~(size_t)255;
                                    return p; };
    const size_t ACT = (size_t)MROWS * DD;        // 4M elems
    us* x  = take(ACT);
    us* qv = take(ACT);
    us* kv = take(ACT);
    us* vv = take(ACT);
    us* t0 = take(ACT);
    us* t1 = take(ACT);
    us* hb = take((size_t)MROWS * DFFD);          // 16M elems
    const size_t WSQ = (size_t)LL * DD * DD;      // 6.29M
    const size_t WSF = (size_t)LL * DD * DFFD;    // 25.2M
    us* wqT = take(WSQ);
    us* wkT = take(WSQ);
    us* wvT = take(WSQ);
    us* woT = take(WSQ);
    us* w1T = take(WSF);
    us* w2T = take(WSF);

    // Convert + transpose all weights fp32 [K][N] -> bf16 [N][K], per call.
    {
        dim3 gq(DD / 32, DD / 32, LL);
        cvt_tr_kernel<<<gq, 256, 0, stream>>>(Wq, wqT, DD, DD);
        cvt_tr_kernel<<<gq, 256, 0, stream>>>(Wk, wkT, DD, DD);
        cvt_tr_kernel<<<gq, 256, 0, stream>>>(Wv, wvT, DD, DD);
        cvt_tr_kernel<<<gq, 256, 0, stream>>>(Wo, woT, DD, DD);
        dim3 g1d(DFFD / 32, DD / 32, LL);         // W1 [D][DFF] -> [DFF][D]
        cvt_tr_kernel<<<g1d, 256, 0, stream>>>(W1, w1T, DD, DFFD);
        dim3 g2d(DD / 32, DFFD / 32, LL);         // W2 [DFF][D] -> [D][DFF]
        cvt_tr_kernel<<<g2d, 256, 0, stream>>>(W2, w2T, DFFD, DD);
    }

    embed_kernel<<<MROWS, 256, 0, stream>>>(tokens, emb, x);

    for (int i = 0; i < LL; ++i) {
        const us* Wq_i = wqT + (size_t)i * DD * DD;
        const us* Wk_i = wkT + (size_t)i * DD * DD;
        const us* Wv_i = wvT + (size_t)i * DD * DD;
        const us* Wo_i = woT + (size_t)i * DD * DD;
        const us* W1_i = w1T + (size_t)i * DD * DFFD;
        const us* W2_i = w2T + (size_t)i * DFFD * DD;

        launch_gemm(x, Wq_i, bq + i * DD, nullptr, qv, MROWS, DD, DD, 0, stream);
        launch_gemm(x, Wk_i, bk + i * DD, nullptr, kv, MROWS, DD, DD, 0, stream);
        launch_gemm(x, Wv_i, bv + i * DD, nullptr, vv, MROWS, DD, DD, 0, stream);

        attention_kernel<<<dim3(SS / 128, HH, BB), 256, 0, stream>>>(
            qv, kv, vv, mask, t0);

        // attn_out + residual(x)
        launch_gemm(t0, Wo_i, bo + i * DD, x, t1, MROWS, DD, DD, 0, stream);
        // out1 = LN(t1) -> reuse qv
        layernorm_kernel<<<MROWS, 256, 0, stream>>>(t1, g1 + i * DD,
                                                    beta1 + i * DD, qv, nullptr);
        // h = relu(out1*W1 + b1)
        launch_gemm(qv, W1_i, b1 + i * DFFD, nullptr, hb, MROWS, DFFD, DD, 1, stream);
        // ffn + residual(out1)
        launch_gemm(hb, W2_i, b2 + i * DD, qv, t0, MROWS, DD, DFFD, 0, stream);
        // x = LN(t0); final layer -> fp32 d_out
        if (i == LL - 1)
            layernorm_kernel<<<MROWS, 256, 0, stream>>>(t0, g2 + i * DD,
                beta2 + i * DD, nullptr, (float*)d_out);
        else
            layernorm_kernel<<<MROWS, 256, 0, stream>>>(t0, g2 + i * DD,
                beta2 + i * DD, x, nullptr);
    }
}
